// SelfAttention_90640989815357
// MI455X (gfx1250) — matrix operand order repaired and checked by reading, not compiled, not run on hardware
//
#include <hip/hip_runtime.h>
#include <hip/hip_bf16.h>

// ---------------- problem constants ----------------
#define HN   16      // heads
#define DH   64      // head dim
#define DM   1024    // model dim
#define BB   2       // batch
#define SS   2048    // seq len
#define MM   (BB*SS) // 4096 rows

typedef __bf16 bf16_t;
typedef __attribute__((ext_vector_type(16))) __bf16 v16bf;
typedef __attribute__((ext_vector_type(8)))  float  v8f;
typedef __attribute__((ext_vector_type(4)))  unsigned int u32x4;
typedef __attribute__((ext_vector_type(8)))  int i32x8;
typedef __attribute__((ext_vector_type(4)))  int i32x4;

typedef __attribute__((address_space(3))) char lds_char;
typedef __attribute__((address_space(1))) char glb_char;

#define HAVE_TDM   __has_builtin(__builtin_amdgcn_tensor_load_to_lds)
#define HAVE_ASYNC __has_builtin(__builtin_amdgcn_global_load_async_to_lds_b128)

#if HAVE_TDM
#  if __has_include(<hip/amd_detail/amd_gfx1250_TDM.h>)
#    define TDM_6ARG 1
#  else
#    define TDM_6ARG 0
#  endif
#endif

union ABf { v16bf v; uint4 q[2]; };

__device__ __forceinline__ unsigned lds_addr_of(void* p) {
    return (unsigned)(unsigned long long)(lds_char*)p;   // AS3 offset
}

// ---------------- conversion kernels ----------------
__global__ void cvt_x_kernel(const float* __restrict__ x, bf16_t* __restrict__ xb, int n) {
    int i = blockIdx.x * 256 + threadIdx.x;
    if (i < n) xb[i] = (bf16_t)x[i];
}

// W[k][n] -> Wt[n][k] bf16; grid.z selects q/k/v
__global__ void cvt_wt_kernel(const float* __restrict__ wq, const float* __restrict__ wk,
                              const float* __restrict__ wv, bf16_t* __restrict__ wt_base) {
    const float* w = (blockIdx.z == 0) ? wq : (blockIdx.z == 1 ? wk : wv);
    bf16_t* wt = wt_base + (size_t)blockIdx.z * DM * DM;
    int i = blockIdx.x * 256 + threadIdx.x;
    int k = i >> 10;
    int n = i & (DM - 1);
    wt[n * DM + k] = (bf16_t)w[i];
}

// ---------------- fused QKV projection GEMM ----------------
// Wave computes 32(M) x 64(N); block = 4 waves covers 32 x 256.
// grid = (MM/32, DM/256, 3). Output bf16 [3][B,H,S,DH].
__global__ __launch_bounds__(128)
void qkv_gemm_kernel(const bf16_t* __restrict__ X, const bf16_t* __restrict__ Wt_base,
                     const float* __restrict__ bq, const float* __restrict__ bk,
                     const float* __restrict__ bv, bf16_t* __restrict__ Y_base) {
    const int lane = threadIdx.x & 31;
    const int wave = threadIdx.x >> 5;
    const int l16  = lane & 15;
    const int half = lane >> 4;

    const float*  bias = (blockIdx.z == 0) ? bq : (blockIdx.z == 1 ? bk : bv);
    const bf16_t* Wt   = Wt_base + (size_t)blockIdx.z * DM * DM;
    bf16_t*       Y    = Y_base  + (size_t)blockIdx.z * MM * DM;

    const int m0 = blockIdx.x * 32;
    const int n0 = blockIdx.y * 256 + wave * 64;

    v8f acc[2][4] = {};

    const bf16_t* xrow0 = X + (size_t)(m0 + l16) * DM;
    const bf16_t* xrow1 = X + (size_t)(m0 + 16 + l16) * DM;

    for (int kk = 0; kk < DM; kk += 32) {
        ABf a0, a1;                       // A frags for the two 16-row sub-tiles
        const int ak = kk + (half ? 16 : 0);
        a0.q[0] = *(const uint4*)(xrow0 + ak);
        a0.q[1] = *(const uint4*)(xrow0 + ak + 8);
        a1.q[0] = *(const uint4*)(xrow1 + ak);
        a1.q[1] = *(const uint4*)(xrow1 + ak + 8);
#pragma unroll
        for (int j = 0; j < 4; ++j) {
            ABf b;                        // B frag reused by both M sub-tiles
            const bf16_t* wcol = Wt + (size_t)(n0 + j * 16 + l16) * DM + kk + (half ? 16 : 0);
            b.q[0] = *(const uint4*)(wcol);
            b.q[1] = *(const uint4*)(wcol + 8);
            acc[0][j] = __builtin_amdgcn_wmma_f32_16x16x32_bf16(
                false, a0.v, false, b.v, (short)0, acc[0][j], false, false);
            acc[1][j] = __builtin_amdgcn_wmma_f32_16x16x32_bf16(
                false, a1.v, false, b.v, (short)0, acc[1][j], false, false);
        }
    }

#pragma unroll
    for (int mi = 0; mi < 2; ++mi)
#pragma unroll
        for (int j = 0; j < 4; ++j) {
            const int n  = n0 + j * 16 + l16;
            const float bvs = bias[n];
            const int h = n >> 6;
            const int d = n & 63;
#pragma unroll
            for (int r = 0; r < 8; ++r) {
                const int m  = m0 + mi * 16 + r + 8 * half;
                const int bi = m >> 11;
                const int s  = m & (SS - 1);
                Y[(((size_t)(bi * HN + h)) * SS + s) * DH + d] = (bf16_t)(acc[mi][j][r] + bvs);
            }
        }
}

// ---------------- K/V tile staging (TDM / async / sync) ----------------
__device__ __forceinline__
void stage_kv(const bf16_t* __restrict__ Kb, const bf16_t* __restrict__ Vb,
              int kt, bf16_t* sk, bf16_t* svt, int tid) {
    // V tile: global -> reg -> transposed LDS stores (overlaps with K DMA)
#pragma unroll
    for (int it = 0; it < 4; ++it) {
        const int unit = tid + it * 128;
        const int row  = unit >> 3;
        const int c8   = (unit & 7) * 8;
        uint4 vv = *(const uint4*)(Vb + (size_t)(kt * 64 + row) * DH + c8);
        const bf16_t* pv = (const bf16_t*)&vv;
#pragma unroll
        for (int e = 0; e < 8; ++e) svt[(c8 + e) * 64 + row] = pv[e];
    }

#if HAVE_TDM
    // K tile via Tensor Data Mover: one DMA per tile, issued by wave 0.
    if ((tid >> 5) == 0) {
        unsigned long long ga = (unsigned long long)(const glb_char*)(const void*)
                                (Kb + (size_t)kt * 64 * DH);
        u32x4 g0;
        g0[0] = 1u;                                   // count=1, user mode
        g0[1] = lds_addr_of(sk);                      // lds_addr
        g0[2] = (unsigned)ga;                         // global_addr[31:0]
        g0[3] = (unsigned)((ga >> 32) & 0x01FFFFFFu) | (2u << 30);  // addr[56:32] | type=2
        i32x8 g1;
        g1[0] = 0x00010000;                           // data_size = 2 bytes
        g1[1] = 64 << 16;                             // tensor_dim0 = 64
        g1[2] = 64 << 16;                             // tensor_dim1 = 64
        g1[3] = 64 << 16;                             // tile_dim0 = 64
        g1[4] = 64;                                   // tile_dim1 = 64
        g1[5] = 64;                                   // tensor_dim0_stride = 64
        g1[6] = 0;
        g1[7] = 0;
        i32x4 z4 = {0, 0, 0, 0};
#  if TDM_6ARG
        i32x8 z8 = {0, 0, 0, 0, 0, 0, 0, 0};
        __builtin_amdgcn_tensor_load_to_lds(g0, g1, z4, z4, z8, 0);
#  else
        __builtin_amdgcn_tensor_load_to_lds(g0, g1, z4, z4, 0);
#  endif
    }
#elif HAVE_ASYNC
    // K tile via async global->LDS copies (4 x b128 per thread, ASYNCcnt)
#pragma unroll
    for (int it = 0; it < 4; ++it) {
        const int unit = tid + it * 128;
        const int row  = unit >> 3;
        const int c8   = (unit & 7) * 8;
        __builtin_amdgcn_global_load_async_to_lds_b128(
            (glb_char*)(const void*)(Kb + (size_t)(kt * 64 + row) * DH + c8),
            (lds_char*)(void*)(sk + row * 64 + c8), 0, 0);
    }
#else
    // sync fallback
#pragma unroll
    for (int it = 0; it < 4; ++it) {
        const int unit = tid + it * 128;
        const int row  = unit >> 3;
        const int c8   = (unit & 7) * 8;
        *(uint4*)(sk + row * 64 + c8) =
            *(const uint4*)(Kb + (size_t)(kt * 64 + row) * DH + c8);
    }
#endif
}

template <int OUT>   // tiles allowed to remain in flight
__device__ __forceinline__ void wait_kv(int wave) {
#if HAVE_TDM
    if (wave == 0) __builtin_amdgcn_s_wait_tensorcnt((short)OUT);
#elif HAVE_ASYNC
#  if __has_builtin(__builtin_amdgcn_s_wait_asynccnt)
    __builtin_amdgcn_s_wait_asynccnt((short)(OUT * 4));
#  else
    if (OUT) asm volatile("s_wait_asynccnt 4" ::: "memory");
    else     asm volatile("s_wait_asynccnt 0" ::: "memory");
#  endif
#else
    (void)wave;
#endif
}

// ---------------- flash attention ----------------
// grid = (SS/64 query tiles, B*H). Block: 4 waves, wave w owns 16 query rows.
// Double-buffered 64x64 K (row-major) and V^T (d-major) tiles in LDS.
__global__ __launch_bounds__(128)
void attn_kernel(const bf16_t* __restrict__ Q, const bf16_t* __restrict__ K,
                 const bf16_t* __restrict__ V, float* __restrict__ out) {
    __shared__ __align__(16) bf16_t sK [2][64 * 64];
    __shared__ __align__(16) bf16_t sVt[2][64 * 64];
    __shared__ __align__(16) bf16_t sP [4][16 * 64];

    const int tid  = threadIdx.x;
    const int lane = tid & 31;
    const int wave = tid >> 5;
    const int l16  = lane & 15;
    const int half = lane >> 4;

    const int qt = blockIdx.x;
    const int bh = blockIdx.y;

    const bf16_t* Qb = Q + (size_t)bh * SS * DH;
    const bf16_t* Kb = K + (size_t)bh * SS * DH;
    const bf16_t* Vb = V + (size_t)bh * SS * DH;

    const int qrow0 = qt * 64 + wave * 16;

    // Q A-frags (2 K-steps over DH=64) held in registers
    ABf qa[2];
    {
        const bf16_t* qr = Qb + (size_t)(qrow0 + l16) * DH;
#pragma unroll
        for (int kk = 0; kk < 2; ++kk) {
            const int ak = kk * 32 + (half ? 16 : 0);
            qa[kk].q[0] = *(const uint4*)(qr + ak);
            qa[kk].q[1] = *(const uint4*)(qr + ak + 8);
        }
    }

    v8f   oacc[4] = {};
    float mrow[8], lrow[8];
#pragma unroll
    for (int r = 0; r < 8; ++r) { mrow[r] = -3.0e38f; lrow[r] = 0.0f; }

    stage_kv(Kb, Vb, 0, sK[0], sVt[0], tid);   // prologue: tile 0 in flight

    for (int kt = 0; kt <= qt; ++kt) {
        const int cur = kt & 1;
        __syncthreads();                       // prior compute done; buffers reusable
        if (kt < qt) {
            stage_kv(Kb, Vb, kt + 1, sK[cur ^ 1], sVt[cur ^ 1], tid);
            wait_kv<1>(wave);                  // tile kt done; kt+1 may stay in flight
        } else {
            wait_kv<0>(wave);
        }
        __syncthreads();                       // tile kt visible to all waves

        const bf16_t* sk  = sK[cur];
        const bf16_t* svt = sVt[cur];

        // S = Q K^T (16 x 64)
        v8f sacc[4] = {};
#pragma unroll
        for (int kk = 0; kk < 2; ++kk) {
#pragma unroll
            for (int j = 0; j < 4; ++j) {
                ABf b;   // B[k=d][n=key]: lane = key, contiguous d run
                const bf16_t* kp = sk + (j * 16 + l16) * 64 + kk * 32 + (half ? 16 : 0);
                b.q[0] = *(const uint4*)kp;
                b.q[1] = *(const uint4*)(kp + 8);
                sacc[j] = __builtin_amdgcn_wmma_f32_16x16x32_bf16(
                    false, qa[kk].v, false, b.v, (short)0, sacc[j], false, false);
            }
        }

        // scale + causal mask (diagonal tile only)
        const bool diag = (kt == qt);
#pragma unroll
        for (int j = 0; j < 4; ++j)
#pragma unroll
            for (int r = 0; r < 8; ++r) {
                float s = sacc[j][r] * 0.125f;
                if (diag) {
                    const int qrow = qrow0 + r + 8 * half;
                    const int kcol = kt * 64 + j * 16 + l16;
                    if (kcol > qrow) s = -1.0e5f;
                }
                sacc[j][r] = s;
            }

        // online softmax: row r lives on one 16-lane half per acc VGPR
#pragma unroll
        for (int r = 0; r < 8; ++r) {
            float vmax = fmaxf(fmaxf(sacc[0][r], sacc[1][r]), fmaxf(sacc[2][r], sacc[3][r]));
#pragma unroll
            for (int mk = 1; mk < 16; mk <<= 1) vmax = fmaxf(vmax, __shfl_xor(vmax, mk, 32));
            const float mnew  = fmaxf(mrow[r], vmax);
            const float alpha = __expf(mrow[r] - mnew);
            mrow[r] = mnew;
            float rs = 0.0f;
#pragma unroll
            for (int j = 0; j < 4; ++j) {
                const float p = __expf(sacc[j][r] - mnew);
                sacc[j][r] = p;
                rs += p;
            }
#pragma unroll
            for (int mk = 1; mk < 16; mk <<= 1) rs += __shfl_xor(rs, mk, 32);
            lrow[r] = lrow[r] * alpha + rs;
#pragma unroll
            for (int j = 0; j < 4; ++j) oacc[j][r] *= alpha;
        }

        // P: C layout -> per-wave LDS -> A layout
        bf16_t* P = sP[wave];
#pragma unroll
        for (int j = 0; j < 4; ++j)
#pragma unroll
            for (int r = 0; r < 8; ++r) {
                const int m = r + 8 * half;
                P[m * 64 + j * 16 + l16] = (bf16_t)sacc[j][r];
            }
        asm volatile("s_wait_dscnt 0" ::: "memory");   // wave-local LDS RAW fence

        // O += P V
#pragma unroll
        for (int kk = 0; kk < 2; ++kk) {
            ABf pa;
            const bf16_t* pr = P + l16 * 64 + kk * 32 + (half ? 16 : 0);
            pa.q[0] = *(const uint4*)pr;
            pa.q[1] = *(const uint4*)(pr + 8);
#pragma unroll
            for (int j = 0; j < 4; ++j) {
                ABf b;   // B[k=key][n=d]: lane = d, contiguous key run
                const bf16_t* vp = svt + (j * 16 + l16) * 64 + kk * 32 + (half ? 16 : 0);
                b.q[0] = *(const uint4*)vp;
                b.q[1] = *(const uint4*)(vp + 8);
                oacc[j] = __builtin_amdgcn_wmma_f32_16x16x32_bf16(
                    false, pa.v, false, b.v, (short)0, oacc[j], false, false);
            }
        }
    }

    // epilogue: O /= l, merge heads -> [B,S,D] f32
    const int bi = bh >> 4;
    const int h  = bh & (HN - 1);
#pragma unroll
    for (int j = 0; j < 4; ++j)
#pragma unroll
        for (int r = 0; r < 8; ++r) {
            const int m = r + 8 * half;
            const int q = qrow0 + m;
            const int d = j * 16 + l16;
            out[((size_t)bi * SS + q) * DM + h * DH + d] = oacc[j][r] / lrow[r];
        }
}

// ---------------- launcher ----------------
extern "C" void kernel_launch(void* const* d_in, const int* in_sizes, int n_in,
                              void* d_out, int out_size, void* d_ws, size_t ws_size,
                              hipStream_t stream) {
    const float* X  = (const float*)d_in[0];
    const float* Wq = (const float*)d_in[1];
    const float* bq = (const float*)d_in[2];
    const float* Wk = (const float*)d_in[3];
    const float* bk = (const float*)d_in[4];
    const float* Wv = (const float*)d_in[5];
    const float* bv = (const float*)d_in[6];
    float* out = (float*)d_out;

    char* ws = (char*)d_ws;
    bf16_t* Xb      = (bf16_t*)ws;   ws += (size_t)MM * DM * 2;
    bf16_t* Wt_base = (bf16_t*)ws;   ws += (size_t)3 * DM * DM * 2;   // Wt q,k,v contiguous
    bf16_t* QKV     = (bf16_t*)ws;   ws += (size_t)3 * MM * DM * 2;   // Q,K,V contiguous
    bf16_t* Qb = QKV;
    bf16_t* Kb = QKV + (size_t)MM * DM;
    bf16_t* Vb = QKV + (size_t)2 * MM * DM;

    // 1) precision conversion (+ weight transpose), fused over q/k/v via grid.z
    cvt_x_kernel<<<(MM * DM + 255) / 256, 256, 0, stream>>>(X, Xb, MM * DM);
    cvt_wt_kernel<<<dim3((DM * DM) / 256, 1, 3), 256, 0, stream>>>(Wq, Wk, Wv, Wt_base);

    // 2) fused QKV projection (WMMA bf16 GEMM)
    qkv_gemm_kernel<<<dim3(MM / 32, DM / 256, 3), 128, 0, stream>>>(
        Xb, Wt_base, bq, bk, bv, QKV);

    // 3) causal flash attention with double-buffered TDM/async K,V tiles
    attn_kernel<<<dim3(SS / 64, BB * HN), 128, 0, stream>>>(Qb, Kb, Vb, out);
}
